// QuantumInspiredLayer_55044300866047
// MI455X (gfx1250) — matrix operand order repaired and checked by reading, not compiled, not run on hardware
//
#include <hip/hip_runtime.h>
#include <hip/hip_bf16.h>
#include <math.h>

// ---------------------------------------------------------------------------
// QuantumInspiredLayer for MI455X (gfx1250, wave32, WMMA + TDM)
//
// Roofline: mandatory HBM traffic ~768MB (x twice + out once) => ~33us @
// 23.3TB/s.  Matrix work ~26 GFLOP after dead-code elimination (only
// W1[:, :64] feeds the output), so bf16 WMMA makes compute negligible.
// W1 operand chunks are staged into LDS by the Tensor Data Mover (wave 0,
// double-buffered, s_wait_tensorcnt) so the K-loop VMEM stream is x only.
// ---------------------------------------------------------------------------

typedef __attribute__((ext_vector_type(16))) __bf16 v16bf;
typedef __attribute__((ext_vector_type(8)))  float  v8f;
typedef __attribute__((ext_vector_type(4)))  unsigned int tdm_g0_t;
typedef __attribute__((ext_vector_type(8)))  int          tdm_g1_t;
typedef __attribute__((ext_vector_type(4)))  int          tdm_g2_t;

#define B_ROWS 16384
#define D_DIM  4096
#define NQ     64
#define QCOLS  64      // only first 64 of the 128 GEMM1 outputs are live
#define LN_EPS 1e-5f

#if defined(__has_builtin)
#if __has_builtin(__builtin_amdgcn_tensor_load_to_lds) && \
    __has_builtin(__builtin_amdgcn_s_wait_tensorcnt)
#define USE_TDM 1
#endif
#endif
#ifndef USE_TDM
#define USE_TDM 0
#endif

static __device__ __forceinline__ v8f wmma_bf16(v16bf a, v16bf b, v8f c) {
  return __builtin_amdgcn_wmma_f32_16x16x32_bf16(
      /*neg_a=*/false, a, /*neg_b=*/false, b,
      /*c_mod=*/(short)0, c, /*reuse_a=*/false, /*reuse_b=*/false);
}

#if USE_TDM
// TDM 1-row copy of `nelem` 16-bit elements from global to LDS.
// D# group0: count=1 | lds_addr | global_addr(57b) | type=2.
// D# group1: data_size=2B, tensor_dim0=tile_dim0=nelem, dims1/2 = 1 row.
static __device__ __forceinline__ void tdm_copy_row_b16(unsigned int lds_off,
                                                        const void* gsrc,
                                                        unsigned int nelem) {
  const unsigned long long ga = (unsigned long long)gsrc;
  tdm_g0_t g0;
  g0[0] = 1u;                                        // count=1, user mode
  g0[1] = lds_off;                                   // lds_addr (bytes)
  g0[2] = (unsigned int)(ga & 0xffffffffu);          // global_addr[31:0]
  g0[3] = (unsigned int)((ga >> 32) & 0x01ffffffu)   // global_addr[56:32]
          | (2u << 30);                              // type=2 ("image")
  tdm_g1_t g1;
  g1[0] = (int)(1u << 16);                           // wg_mask=0, data_size=1 (2B)
  g1[1] = (int)((nelem & 0xffffu) << 16);            // tensor_dim0[15:0]
  g1[2] = (int)((nelem >> 16) | (1u << 16));         // dim0[31:16] | tensor_dim1=1
  g1[3] = (int)((nelem & 0xffffu) << 16);            // dim1 hi=0 | tile_dim0
  g1[4] = 1;                                         // tile_dim1=1, tile_dim2=0
  g1[5] = (int)nelem;                                // tensor_dim0_stride[31:0]
  g1[6] = 0;                                         // stride hi | dim1_stride lo
  g1[7] = 0;
  tdm_g2_t gz = {0, 0, 0, 0};
#if __clang_major__ >= 23
  tdm_g1_t gz8 = {0, 0, 0, 0, 0, 0, 0, 0};
  __builtin_amdgcn_tensor_load_to_lds(g0, g1, gz, gz, gz8, 0);
#else
  __builtin_amdgcn_tensor_load_to_lds(g0, g1, gz, gz, 0);
#endif
}
#endif

// ---------------------------------------------------------------------------
// Kernel 0: one-time bf16 repack of W1[:, :64] and W2 into per-lane WMMA
// B-operand layout: [Kchunk][colTile][lane 0..31][e 0..15], where in W2p lane<16
// holds K=0..15 (N=lane) and lane>=16 holds K=16..31 (N=lane-16); W1p: K=8h+e if e<8 else 8+8h+e (h=lane/16).
// ---------------------------------------------------------------------------
__global__ void pack_weights(const float* __restrict__ W1,
                             const float* __restrict__ W2,
                             __bf16* __restrict__ W1p,   // [128][4][32][16]
                             __bf16* __restrict__ W2p) { // [4][256][32][16]
  const int tid = blockIdx.x * blockDim.x + threadIdx.x;
  const int nthreads = gridDim.x * blockDim.x;
  for (int i = tid; i < 128 * 4 * 32 * 16; i += nthreads) {
    const int e    = i & 15;
    const int lane = (i >> 4) & 31;
    const int t    = (i >> 9) & 3;
    const int c    = i >> 11;
    const int kl   = (e < 8) ? ((lane >> 4) * 8 + e) : (16 + (lane >> 4) * 8 + (e - 8));
    const int K    = c * 32 + kl;
    const int N    = t * 16 + (lane & 15);
    W1p[i] = (__bf16)W1[(size_t)K * (2 * NQ) + N];   // W1 is [4096][128]
  }
  for (int i = tid; i < 4 * 256 * 32 * 16; i += nthreads) {
    const int e    = i & 15;
    const int lane = (i >> 4) & 31;
    const int t    = (i >> 9) & 255;
    const int c    = i >> 17;
    const int kl   = (lane < 16) ? e : (16 + e);
    const int K    = c * 32 + kl;
    const int N    = t * 16 + (lane & 15);
    W2p[i] = (__bf16)W2[(size_t)K * D_DIM + N];      // W2 is [128][4096]
  }
}

// ---------------------------------------------------------------------------
// Kernel 1: q = tanh(x @ W1[:, :64] + b1[:64]).
// 256 blocks x 4 waves; wave w owns rows [blk*64 + 16w, +16) and all 4
// column tiles -> x is read exactly once from HBM.  W1p K-chunks (4KB) are
// shared by all 4 waves: wave 0 TDM-stages them into double-buffered LDS.
// ---------------------------------------------------------------------------
__global__ void __launch_bounds__(128)
gemm1_tanh(const float* __restrict__ x,
           const __bf16* __restrict__ W1p,
           const float* __restrict__ b1,
           float* __restrict__ q) {               // [B][64] fp32
  __shared__ __bf16 sW1[2][4 * 32 * 16];          // 2 x 4KB double buffer
  const int wave    = threadIdx.x >> 5;
  const int lane    = threadIdx.x & 31;
  const int rowBase = blockIdx.x * 64 + wave * 16;
  const int m       = lane & 15;
  const int hi      = lane >> 4;

  const float* xrow = x + (size_t)(rowBase + m) * D_DIM;

#if USE_TDM
  const unsigned int lds0 = (unsigned int)(unsigned long long)(const void*)&sW1[0][0];
  const unsigned int lds1 = (unsigned int)(unsigned long long)(const void*)&sW1[1][0];
  if (threadIdx.x < 32) {                          // leader wave issues DMA
    tdm_copy_row_b16(lds0, W1p, 2048);             // chunk 0 -> buf 0
  }
#endif

  v8f acc0 = {}, acc1 = {}, acc2 = {}, acc3 = {};

#pragma unroll 1
  for (int c = 0; c < 128; ++c) {
    const int buf = c & 1;
#if USE_TDM
    if (threadIdx.x < 32) {
      // Issue next chunk into the other buffer ((c+1)&127 keeps the wait
      // count a constant immediate; the final redundant issue is harmless).
      const int cn = (c + 1) & 127;
      tdm_copy_row_b16(buf ? lds0 : lds1, W1p + (size_t)cn * 2048, 2048);
      __builtin_amdgcn_s_wait_tensorcnt(1);        // chunk c has landed
    }
    __syncthreads();                               // buf[c&1] visible to all
#endif

    const int kbase = c * 32 + hi * 8;
    __builtin_prefetch(xrow + kbase + 64, 0, 1);   // global_prefetch_b8
    const float4 a0 = *(const float4*)(xrow + kbase);
    const float4 a1 = *(const float4*)(xrow + kbase + 4);
    const float4 a2 = *(const float4*)(xrow + kbase + 16);
    const float4 a3 = *(const float4*)(xrow + kbase + 20);
    v16bf a;
    a[0]  = (__bf16)a0.x; a[1]  = (__bf16)a0.y; a[2]  = (__bf16)a0.z; a[3]  = (__bf16)a0.w;
    a[4]  = (__bf16)a1.x; a[5]  = (__bf16)a1.y; a[6]  = (__bf16)a1.z; a[7]  = (__bf16)a1.w;
    a[8]  = (__bf16)a2.x; a[9]  = (__bf16)a2.y; a[10] = (__bf16)a2.z; a[11] = (__bf16)a2.w;
    a[12] = (__bf16)a3.x; a[13] = (__bf16)a3.y; a[14] = (__bf16)a3.z; a[15] = (__bf16)a3.w;

#if USE_TDM
    const __bf16* bp = &sW1[buf][lane * 16];       // ds_load_b128 pairs
#else
    const __bf16* bp = W1p + ((size_t)(c * 4) * 32 + lane) * 16;
#endif
    const v16bf b0 = *(const v16bf*)(bp);
    const v16bf b1v = *(const v16bf*)(bp + 512);
    const v16bf b2v = *(const v16bf*)(bp + 1024);
    const v16bf b3v = *(const v16bf*)(bp + 1536);

    acc0 = wmma_bf16(a, b0, acc0);
    acc1 = wmma_bf16(a, b1v, acc1);
    acc2 = wmma_bf16(a, b2v, acc2);
    acc3 = wmma_bf16(a, b3v, acc3);

#if USE_TDM
    __syncthreads();   // nobody still reads buf before TDM overwrites it
#endif
  }

  // Epilogue: +b1, tanh, scatter to q[B,64].
  // C layout: VGPR r -> (M=r, N=lane) for lanes 0-15; (M=r+8, N=lane-16) else.
  v8f accs[4] = {acc0, acc1, acc2, acc3};
#pragma unroll
  for (int t = 0; t < 4; ++t) {
    const float bb = b1[t * 16 + m];
#pragma unroll
    for (int r = 0; r < 8; ++r) {
      const int rowM = hi ? (r + 8) : r;
      const float v = tanhf(accs[t][r] + bb);
      q[(size_t)(rowBase + rowM) * QCOLS + t * 16 + m] = v;
    }
  }
}

// ---------------------------------------------------------------------------
// Kernel 2: streaming quantum feature scan, one thread per row.
// Only qubits 0..31 reach the output; the swap condition propagates forward
// one qubit at a time, so O(1) state suffices.  Trig args are bounded by pi,
// so hardware __sinf/__cosf are accurate.  Emits bf16 features pre-packed in
// the WMMA-A layout for GEMM2: [rowTile][Kchunk][laneSlot][e].
// ---------------------------------------------------------------------------
__global__ void quantum_feats(const float* __restrict__ q,
                              __bf16* __restrict__ Fp) { // [1024][4][32][16]
  const int row = blockIdx.x * blockDim.x + threadIdx.x;
  if (row >= B_ROWS) return;
  const float* qr = q + (size_t)row * QCOLS;
  const float inv_sqrt2 = 0.70710678118654752440f;
  const float PI_F = 3.14159265358979323846f;
  const int rowTile = row >> 4;
  const int mm = row & 15;
  __bf16* base = Fp + (size_t)rowTile * (4 * 32 * 16);

  bool cond = false;
#pragma unroll 1
  for (int i = 0; i < 32; ++i) {
    const float theta = qr[2 * i] * PI_F;
    const float phi   = qr[2 * i + 1] * PI_F;
    const float h = 0.5f * theta;
    float ar = __cosf(h), ai = 0.0f;
    const float s = __sinf(h);
    float br = s * __cosf(phi), bi = s * __sinf(phi);
    if ((i & 1) == 0) {   // Hadamard on even qubits
      const float har = (ar + br) * inv_sqrt2, hai = (ai + bi) * inv_sqrt2;
      const float hbr = (ar - br) * inv_sqrt2, hbi = (ai - bi) * inv_sqrt2;
      ar = har; ai = hai; br = hbr; bi = hbi;
    }
    if (cond) { const float tr = ar, ti = ai; ar = br; ai = bi; br = tr; bi = ti; }
    cond = (br * br + bi * bi) > 0.25f;

    const float fv[4] = {ar, ai, br, bi};
#pragma unroll
    for (int j = 0; j < 4; ++j) {
      const int K = 4 * i + j;
      const int c = K >> 5;
      const int kl = K & 31;
      const int laneSlot = (kl < 16) ? mm : (16 + mm);
      const int e = kl & 15;
      base[((size_t)c * 32 + laneSlot) * 16 + e] = (__bf16)fv[j];
    }
  }
}

// ---------------------------------------------------------------------------
// Kernel 3: out = LN(feats @ W2 + b2) * gamma + beta + x.
// 1024 blocks (16 rows) x 8 waves.  A (feats, K=128) stays in registers;
// waves sweep 256 column tiles (4 wmma each), stage the 16x4096 fp32 block
// in 256KB LDS (<=320KB WGP budget), then do mean/var via wave32 shuffles
// and write the normalized residual output with coalesced stores.
// ---------------------------------------------------------------------------
__global__ void __launch_bounds__(256)
gemm2_ln_residual(const __bf16* __restrict__ Fp,
                  const __bf16* __restrict__ W2p,
                  const float* __restrict__ b2,
                  const float* __restrict__ x,
                  const float* __restrict__ gamma,
                  const float* __restrict__ beta,
                  float* __restrict__ out) {
  extern __shared__ float sout[];               // [16][4096] fp32 = 256 KB
  const int wave    = threadIdx.x >> 5;         // 0..7
  const int lane    = threadIdx.x & 31;
  const int rowTile = blockIdx.x;
  const int rowBase = rowTile * 16;
  const int m       = lane & 15;
  const int hi      = lane >> 4;

  const __bf16* ab = Fp + (size_t)rowTile * (4 * 32 * 16);
  const v16bf A0 = *(const v16bf*)(ab + (0 * 32 + lane) * 16);
  const v16bf A1 = *(const v16bf*)(ab + (1 * 32 + lane) * 16);
  const v16bf A2 = *(const v16bf*)(ab + (2 * 32 + lane) * 16);
  const v16bf A3 = *(const v16bf*)(ab + (3 * 32 + lane) * 16);

#pragma unroll 1
  for (int t = wave; t < 256; t += 8) {
    const __bf16* bp = W2p + ((size_t)t * 32 + lane) * 16;  // chunk stride 131072
    __builtin_prefetch(bp + 8 * 512, 0, 1);                 // next tile of wave
    const v16bf B0 = *(const v16bf*)(bp);
    const v16bf B1 = *(const v16bf*)(bp + 131072);
    const v16bf B2 = *(const v16bf*)(bp + 2 * 131072);
    const v16bf B3 = *(const v16bf*)(bp + 3 * 131072);

    v8f acc = {};
    acc = wmma_bf16(A0, B0, acc);
    acc = wmma_bf16(A1, B1, acc);
    acc = wmma_bf16(A2, B2, acc);
    acc = wmma_bf16(A3, B3, acc);

    const float bb = b2[t * 16 + m];
#pragma unroll
    for (int r = 0; r < 8; ++r) {
      const int rowM = hi ? (r + 8) : r;
      sout[rowM * D_DIM + t * 16 + m] = acc[r] + bb;
    }
  }
  __syncthreads();

  // LayerNorm + residual: each wave owns 2 rows, lanes stride the 4096 cols.
#pragma unroll 1
  for (int rr = 0; rr < 2; ++rr) {
    const int row = wave * 2 + rr;
    float sum = 0.0f, sumsq = 0.0f;
#pragma unroll 1
    for (int cc = 0; cc < 128; ++cc) {
      const float v = sout[row * D_DIM + cc * 32 + lane];
      sum += v; sumsq += v * v;
    }
#pragma unroll
    for (int off = 16; off > 0; off >>= 1) {
      sum   += __shfl_xor(sum, off, 32);
      sumsq += __shfl_xor(sumsq, off, 32);
    }
    const float mu   = sum * (1.0f / (float)D_DIM);
    const float var  = sumsq * (1.0f / (float)D_DIM) - mu * mu;
    const float rstd = rsqrtf(var + LN_EPS);
    const float* xr = x + (size_t)(rowBase + row) * D_DIM;
    float* orow = out + (size_t)(rowBase + row) * D_DIM;
#pragma unroll 1
    for (int cc = 0; cc < 128; ++cc) {
      const int col = cc * 32 + lane;
      const float v = sout[row * D_DIM + col];
      orow[col] = (v - mu) * rstd * gamma[col] + beta[col] + xr[col];
    }
  }
}

// ---------------------------------------------------------------------------
extern "C" void kernel_launch(void* const* d_in, const int* in_sizes, int n_in,
                              void* d_out, int out_size, void* d_ws, size_t ws_size,
                              hipStream_t stream) {
  const float* x     = (const float*)d_in[0];
  const float* W1    = (const float*)d_in[1];
  const float* b1    = (const float*)d_in[2];
  const float* W2    = (const float*)d_in[3];
  const float* b2    = (const float*)d_in[4];
  const float* gam   = (const float*)d_in[5];
  const float* bet   = (const float*)d_in[6];
  float* out = (float*)d_out;

  char* ws = (char*)d_ws;
  __bf16* W1p = (__bf16*)ws;                                  // 512 KB
  __bf16* W2p = (__bf16*)(ws + (size_t)(512 << 10));          // 1 MB
  float*  q   = (float*) (ws + (size_t)(1536 << 10));         // 4 MB
  __bf16* Fp  = (__bf16*)(ws + (size_t)(1536 << 10) + ((size_t)4 << 20)); // 4 MB

  pack_weights<<<512, 256, 0, stream>>>(W1, W2, W1p, W2p);
  gemm1_tanh<<<B_ROWS / 64, 128, 0, stream>>>(x, W1p, b1, q);
  quantum_feats<<<B_ROWS / 256, 256, 0, stream>>>(q, Fp);
  gemm2_ln_residual<<<B_ROWS / 16, 256, 16 * D_DIM * sizeof(float), stream>>>(
      Fp, W2p, b2, x, gam, bet, out);
}
